// HME_1271310320153
// MI455X (gfx1250) — matrix operand order repaired and checked by reading, not compiled, not run on hardware
//
#include <hip/hip_runtime.h>
#include <hip/hip_bf16.h>

typedef __attribute__((ext_vector_type(16))) __bf16 v16bf;
typedef __attribute__((ext_vector_type(8)))  __bf16 v8bf;
typedef __attribute__((ext_vector_type(8)))  float  v8f;

#define DEPTH   6
#define NLEAF   64
#define GC      63
#define BATCH   1024
#define GF      512
#define INF     512
#define OUTF    512
#define KTOT    (INF * NLEAF)   /* 32768 */
#define NCHUNK  (KTOT / 32)     /* 1024  */

__device__ __forceinline__ __bf16 f2bf(float f) { return (__bf16)f; }

/* In-place accumulating WMMA with the accumulator PINNED to a fixed
   physical VGPR tuple. */
__device__ __forceinline__ void wmma_acc0(v8f& acc, const v16bf& a, const v16bf& b) {
    asm("v_wmma_f32_16x16x32_bf16 %0, %1, %2, %0"
        : "+{v[40:47]}"(acc)
        : "v"(a), "v"(b));
}
__device__ __forceinline__ void wmma_acc1(v8f& acc, const v16bf& a, const v16bf& b) {
    asm("v_wmma_f32_16x16x32_bf16 %0, %1, %2, %0"
        : "+{v[48:55]}"(acc)
        : "v"(a), "v"(b));
}
/* zero-instruction anchors: force the value to sit in the pinned tuple
   at block boundaries so no boundary copies are ever profitable */
__device__ __forceinline__ void pin_acc(v8f& c0, v8f& c1) {
    asm("" : "+{v[40:47]}"(c0));
    asm("" : "+{v[48:55]}"(c1));
}

/* ------------------------------------------------------------------ *
 * Kernel 1: gating GEMV + sigmoid + soft-decision-tree leaf probs.   *
 * ------------------------------------------------------------------ */
__global__ void __launch_bounds__(64)
leafprob_kernel(const float* __restrict__ xg, const float* __restrict__ gw,
                const float* __restrict__ gb, float* __restrict__ leafp)
{
    const int b = blockIdx.x;
    const int j = threadIdx.x;
    __shared__ float g[NLEAF];

    const float* xrow = xg + (size_t)b * GF;
    if (j < GC) {
        float acc = 0.f;
        #pragma unroll 4
        for (int k = 0; k < GF; ++k)
            acc += xrow[k] * gw[(size_t)k * GC + j];
        acc += gb[j];
        g[j] = 1.f / (1.f + __expf(-acc));
    }
    __syncthreads();

    const int l = j;
    float p = 1.f;
    #pragma unroll
    for (int d = 0; d < DEPTH; ++d) {
        int shift = DEPTH - d;
        int par   = l >> shift;
        int bit   = (l >> (shift - 1)) & 1;
        float gg  = g[(1 << d) - 1 + par];
        p *= bit ? (1.f - gg) : gg;
    }
    leafp[(size_t)b * NLEAF + l] = p;
}

/* ------------------------------------------------------------------ *
 * Kernel 1b: one-time pw f32 -> bf16 conversion into workspace.      *
 * ------------------------------------------------------------------ */
__global__ void __launch_bounds__(256)
pw_cvt_kernel(const float4* __restrict__ in, v8bf* __restrict__ outb)
{
    size_t t = (size_t)blockIdx.x * 256 + threadIdx.x;   /* one per 8 elems */
    float4 a = in[2 * t];
    float4 b = in[2 * t + 1];
    v8bf r;
    r[0] = f2bf(a.x); r[1] = f2bf(a.y); r[2] = f2bf(a.z); r[3] = f2bf(a.w);
    r[4] = f2bf(b.x); r[5] = f2bf(b.y); r[6] = f2bf(b.z); r[7] = f2bf(b.w);
    outb[t] = r;
}

/* ================== shared GEMM helpers (device) =================== */

__device__ __forceinline__ void stage_lp(float* lp_t, const float* __restrict__ leafp,
                                         int m0, int tid)
{
    for (int idx = tid; idx < 64 * NLEAF; idx += 256) {
        int m = idx >> 6;
        int l = idx & 63;
        lp_t[l * 64 + m] = leafp[(size_t)(m0 + m) * NLEAF + l];
    }
    __syncthreads();
}

__device__ __forceinline__ void build_fragL(v16bf fragL[2], const float* lp_t,
                                            int am, int hl)
{
    #pragma unroll
    for (int h = 0; h < 2; ++h) {
        int lb = h * 32 + hl * 16;
        #pragma unroll
        for (int e = 0; e < 8; ++e) {
            fragL[h][e]     = f2bf(lp_t[(lb + e) * 64 + am]);
            fragL[h][e + 8] = f2bf(lp_t[(lb + 8 + e) * 64 + am]);
        }
    }
}

__device__ __forceinline__ void epilogue_store(v8f& c0, v8f& c1, const float* lp_t,
                                               const float* __restrict__ pb,
                                               float* __restrict__ out,
                                               int m0, int o0, int mbase)
{
    const float* pbc0 = pb + (size_t)o0 * NLEAF;
    const float* pbc1 = pbc0 + (size_t)16 * NLEAF;
    #pragma unroll 4
    for (int l = 0; l < NLEAF; ++l) {
        const float* lrow = &lp_t[l * 64 + mbase];
        float pv0 = pbc0[l];
        float pv1 = pbc1[l];
        #pragma unroll
        for (int v = 0; v < 8; ++v) {
            float lv = lrow[v];
            c0[v] += lv * pv0;
            c1[v] += lv * pv1;
        }
    }
    #pragma unroll
    for (int v = 0; v < 8; ++v) {
        int r = m0 + mbase + v;
        out[(size_t)r * OUTF + o0]      = c0[v];
        out[(size_t)r * OUTF + o0 + 16] = c1[v];
    }
}

/* ------------------------------------------------------------------ *
 * Kernel 2a (preferred): GEMM over pre-converted bf16 pw.            *
 * Single do-while over all chunks (wrapped 2-ahead prefetch, no tail)*
 * + pinned accumulators anchored at every block boundary.            *
 * ------------------------------------------------------------------ */
__global__ void __launch_bounds__(256)
moe_wmma_bf16_kernel(const float* __restrict__ xl, const __bf16* __restrict__ pwb,
                     const float* __restrict__ pb, const float* __restrict__ leafp,
                     float* __restrict__ out)
{
    __shared__ float lp_t[NLEAF * 64];

    const int m0  = blockIdx.x * 64;
    const int oc0 = blockIdx.y * 64;
    const int tid = threadIdx.x;
    const int lane = tid & 31;
    const int w  = tid >> 5;
    const int mw = w & 3;
    const int nw = w >> 2;
    const int hl = lane >> 4;
    const int ln = lane & 15;
    const int am = mw * 16 + ln;

    stage_lp(lp_t, leafp, m0, tid);
    v16bf fragL[2];
    build_fragL(fragL, lp_t, am, hl);

    const float* xrow = xl + (size_t)(m0 + am) * INF;
    const int o0 = oc0 + nw * 32 + ln;
    const __bf16* pwc0 = pwb + (size_t)o0 * KTOT + hl * 16;
    const __bf16* pwc1 = pwc0 + (size_t)16 * KTOT;

    v8f c0 = {}; v8f c1 = {};

    v16bf b0a, b1a, b0b, b1b;
    b0a = *(const v16bf*)(pwc0);
    b1a = *(const v16bf*)(pwc1);

    int c = 0;
    do {
        pin_acc(c0, c1);                     /* 0-instr boundary anchor */

        b0b = *(const v16bf*)(pwc0 + (size_t)(c + 1) * 32);
        b1b = *(const v16bf*)(pwc1 + (size_t)(c + 1) * 32);

        /* chunks c (even) and c+1 (odd) share i = c>>1, hence one x scalar;
           aA/aB built together -> overlapping live ranges -> distinct regs */
        __bf16 xb = f2bf(xrow[c >> 1]);
        v16bf aA = fragL[0] * xb;
        v16bf aB = fragL[1] * xb;

        wmma_acc0(c0, aA, b0a);
        wmma_acc1(c1, aA, b1a);

        /* 2-ahead prefetch; wraps to chunk 0 on the final body (harmless) */
        int cn = (c + 2) & (NCHUNK - 1);
        b0a = *(const v16bf*)(pwc0 + (size_t)cn * 32);
        b1a = *(const v16bf*)(pwc1 + (size_t)cn * 32);

        wmma_acc0(c0, aB, b0b);
        wmma_acc1(c1, aB, b1b);

        c += 2;
    } while (c < NCHUNK);
    pin_acc(c0, c1);

    /* WMMA->VALU RAW-on-D guard before the epilogue reads accumulators */
    asm volatile("v_nop\n\tv_nop\n\tv_nop\n\tv_nop");

    epilogue_store(c0, c1, lp_t, pb, out, m0, o0, mw * 16 + hl * 8);
}

/* ------------------------------------------------------------------ *
 * Kernel 2b (fallback, small workspace): convert pw in the loop.     *
 * (intrinsic WMMA: compiler-managed hazards)                         *
 * ------------------------------------------------------------------ */
__global__ void __launch_bounds__(256)
moe_wmma_f32src_kernel(const float* __restrict__ xl, const float* __restrict__ pw,
                       const float* __restrict__ pb, const float* __restrict__ leafp,
                       float* __restrict__ out)
{
    __shared__ float lp_t[NLEAF * 64];

    const int m0  = blockIdx.x * 64;
    const int oc0 = blockIdx.y * 64;
    const int tid = threadIdx.x;
    const int lane = tid & 31;
    const int w  = tid >> 5;
    const int mw = w & 3;
    const int nw = w >> 2;
    const int hl = lane >> 4;
    const int ln = lane & 15;
    const int am = mw * 16 + ln;

    stage_lp(lp_t, leafp, m0, tid);
    v16bf fragL[2];
    build_fragL(fragL, lp_t, am, hl);

    const float* xrow = xl + (size_t)(m0 + am) * INF;
    const int o0 = oc0 + nw * 32 + ln;
    const float* pwc0 = pw + (size_t)o0 * KTOT + hl * 16;
    const float* pwc1 = pwc0 + (size_t)16 * KTOT;

    v8f c0 = {}; v8f c1 = {};

    #pragma unroll 2
    for (int c = 0; c < NCHUNK; ++c) {
        __bf16 xb = f2bf(xrow[c >> 1]);
        v16bf a = fragL[c & 1] * xb;

        const float4* p0 = (const float4*)(pwc0 + (size_t)c * 32);
        const float4* p1 = (const float4*)(pwc1 + (size_t)c * 32);
        v16bf b0, b1;
        #pragma unroll
        for (int q = 0; q < 4; ++q) {
            float4 f0 = p0[q];
            float4 f1 = p1[q];
            b0[4*q+0] = f2bf(f0.x); b0[4*q+1] = f2bf(f0.y);
            b0[4*q+2] = f2bf(f0.z); b0[4*q+3] = f2bf(f0.w);
            b1[4*q+0] = f2bf(f1.x); b1[4*q+1] = f2bf(f1.y);
            b1[4*q+2] = f2bf(f1.z); b1[4*q+3] = f2bf(f1.w);
        }
        c0 = __builtin_amdgcn_wmma_f32_16x16x32_bf16(false, a, false, b0,
                                                     (short)0, c0, false, false);
        c1 = __builtin_amdgcn_wmma_f32_16x16x32_bf16(false, a, false, b1,
                                                     (short)0, c1, false, false);
    }

    epilogue_store(c0, c1, lp_t, pb, out, m0, o0, mw * 16 + hl * 8);
}

extern "C" void kernel_launch(void* const* d_in, const int* in_sizes, int n_in,
                              void* d_out, int out_size, void* d_ws, size_t ws_size,
                              hipStream_t stream) {
    (void)in_sizes; (void)n_in; (void)out_size;
    const float* xg = (const float*)d_in[0];
    const float* xl = (const float*)d_in[1];
    const float* gw = (const float*)d_in[2];
    const float* gb = (const float*)d_in[3];
    const float* pw = (const float*)d_in[4];
    const float* pb = (const float*)d_in[5];
    float* out   = (float*)d_out;

    const size_t lp_bytes  = (size_t)BATCH * NLEAF * sizeof(float);   /* 256 KB  */
    const size_t pwb_bytes = (size_t)KTOT * OUTF * sizeof(__bf16);    /* 33.5 MB */

    float* leafp = (float*)d_ws;
    leafprob_kernel<<<BATCH, 64, 0, stream>>>(xg, gw, gb, leafp);

    dim3 grid(BATCH / 64, OUTF / 64);   /* 16 x 8 workgroups, 256 thr */

    if (ws_size >= lp_bytes + pwb_bytes) {
        /* preferred: one-time bf16 conversion of pw, then lean GEMM */
        __bf16* pwb = (__bf16*)((char*)d_ws + lp_bytes);
        int nblk = (int)(((size_t)KTOT * OUTF / 8) / 256);            /* 8192 */
        pw_cvt_kernel<<<nblk, 256, 0, stream>>>((const float4*)pw, (v8bf*)pwb);
        moe_wmma_bf16_kernel<<<grid, 256, 0, stream>>>(xl, pwb, pb, leafp, out);
    } else {
        /* fallback: convert pw inside the GEMM inner loop */
        moe_wmma_f32src_kernel<<<grid, 256, 0, stream>>>(xl, pw, pb, leafp, out);
    }
}
